// LinearAttention_89258010345979
// MI455X (gfx1250) — compile-verified, matrix operand order checked by reading, not run on hardware
//
#include <hip/hip_runtime.h>
#include <hip/hip_bf16.h>
#include <math.h>

// ---------------------------------------------------------------------------
// LinearAttention fused pipeline for gfx1250 (MI455X), wave32 + WMMA bf16.
//   x:(16,256,64,64) -> rmsnorm -> qkv GEMM -> softmaxes -> ctx/out GEMMs
//   -> out proj GEMM + bias -> rmsnorm -> d_out
// GEMMs use v_wmma_f32_16x16x32_bf16; the weight (A) tile is staged with the
// Tensor Data Mover (tensor_load_to_lds, double-buffered, s_wait_tensorcnt).
// ---------------------------------------------------------------------------

typedef __bf16 bf16_t;
typedef __attribute__((ext_vector_type(8)))  __bf16 bf16x8;
typedef __attribute__((ext_vector_type(16))) __bf16 bf16x16;
typedef __attribute__((ext_vector_type(8)))  float  floatx8;
typedef __attribute__((ext_vector_type(4)))  unsigned int u32x4;
typedef __attribute__((ext_vector_type(8)))  int i32x8;
typedef __attribute__((ext_vector_type(4)))  int i32x4;

#define NTOK 4096          // 64*64 pixels
#define NBATCH 16
#define IN_CH 256
#define HIDDEN 512
#define O3 1536            // 3*HIDDEN
#define NHEADS 8
#define DHEAD 64

#ifndef __has_builtin
#define __has_builtin(x) 0
#endif
#if defined(__HIP_DEVICE_COMPILE__) && __has_builtin(__builtin_amdgcn_tensor_load_to_lds)
#define HAVE_TDM 1
#else
#define HAVE_TDM 0
#endif

#if HAVE_TDM
// Build a D# (CDNA5 ISA) for a 2D bf16 tile copy
// global->LDS: tile_dim0=tileK contiguous elements, tile_dim1=tileM rows,
// row stride = strideElems. Tensor dims set huge (tile is always in-bounds).
// This toolchain's builtin is the 6-arg form (5 descriptor SGPR groups + cpol).
__device__ __forceinline__ void tdm_load_tile_bf16(unsigned ldsOff, const __bf16* gsrc,
                                                   int tileK, int tileM,
                                                   unsigned strideElems) {
  unsigned long long ga = (unsigned long long)(const void*)gsrc;
  u32x4 g0;
  g0[0] = 1u;                                   // count=1, user mode
  g0[1] = ldsOff;                               // lds_addr (bytes)
  g0[2] = (unsigned)ga;                         // global_addr[31:0]
  g0[3] = (unsigned)((ga >> 32) & 0x01ffffffu)  // global_addr[56:32]
          | (2u << 30);                         // type = 2 ("image")
  i32x8 g1;
  g1[0] = 1 << 16;                              // workgroup_mask=0, data_size=1 (2B)
  g1[1] = (int)0xffff0000u;                     // tensor_dim0[15:0]=0xffff (bits 63:48)
  g1[2] = (int)(0x00007fffu | 0xffff0000u);     // tensor_dim0 hi / tensor_dim1 lo
  g1[3] = (int)(0x00007fffu | ((unsigned)tileK << 16)); // tensor_dim1 hi | tile_dim0
  g1[4] = tileM & 0xffff;                       // tile_dim1 (tile_dim2 = 0)
  g1[5] = (int)strideElems;                     // tensor_dim0_stride[31:0]
  g1[6] = 0;                                    // stride hi / dim1_stride lo
  g1[7] = 0;
  i32x4 z4 = {0, 0, 0, 0};
  i32x8 z8 = {0, 0, 0, 0, 0, 0, 0, 0};
  __builtin_amdgcn_tensor_load_to_lds(g0, g1, z4, z4, z8, 0);
}
#endif

// --- WMMA fragment loader -------------------------------------------------
// LDS tile layout is row-major [row][32 or 64 k-elements].  Per CDNA5 ISA
// (CDNA5 ISA): 16-bit A/B fragment for 16x16x32: lane L holds row (M or N)
// = L%16 and 16 K values = { (L>=16?8:0)+0..7 , 16+(L>=16?8:0)+0..7 }.
// Both runs are 8 contiguous bf16 -> two 16-byte LDS loads.
__device__ __forceinline__ bf16x16 load_frag(const __bf16* rowp, int kh) {
  bf16x8 lo = *(const bf16x8*)(rowp + kh);
  bf16x8 hi = *(const bf16x8*)(rowp + 16 + kh);
  return __builtin_shufflevector(lo, hi, 0,1,2,3,4,5,6,7,8,9,10,11,12,13,14,15);
}

// --- Kernel 1: per-pixel inv-norm folded into bf16 x ----------------------
__global__ __launch_bounds__(256)
void prep_x_kernel(const float* __restrict__ x, __bf16* __restrict__ xh) {
  int gid = blockIdx.x * blockDim.x + threadIdx.x;  // b*4096 + j
  int b = gid >> 12, j = gid & (NTOK - 1);
  const float* xp = x + (size_t)b * IN_CH * NTOK + j;
  float ss = 0.f;
  for (int c = 0; c < IN_CH; ++c) { float v = xp[(size_t)c * NTOK]; ss += v * v; }
  float s = 16.f / fmaxf(sqrtf(ss), 1e-12f);   // sqrt(C)=16 folded here
  __bf16* xo = xh + (size_t)b * IN_CH * NTOK + j;
  for (int c = 0; c < IN_CH; ++c) xo[(size_t)c * NTOK] = (__bf16)(xp[(size_t)c * NTOK] * s);
}

// --- Kernel 2: weights -> bf16 (g1 folded into Wqkv) ----------------------
__global__ __launch_bounds__(256)
void prep_w_kernel(const float* __restrict__ Wqkv, const float* __restrict__ g1,
                   const float* __restrict__ Wout,
                   __bf16* __restrict__ Wh, __bf16* __restrict__ Wouth) {
  int i = blockIdx.x * 256 + threadIdx.x;
  if (i < O3 * IN_CH) Wh[i] = (__bf16)(Wqkv[i] * g1[i & (IN_CH - 1)]);
  int i2 = i - O3 * IN_CH;
  if (i2 >= 0 && i2 < IN_CH * HIDDEN) Wouth[i2] = (__bf16)Wout[i2];
}

// --- Kernel 3/7: bf16 GEMM  C[b][m][n] = A[m][k] * B[b][k][n] (+bias) ----
// Tile: 128(M) x 64(N) x 32(K). 8 waves; wave grid 4x2, each wave 2x2 WMMA.
// A staged by TDM (double-buffered); B loaded b128-coalesced and transposed
// on the LDS-write side.
__global__ __launch_bounds__(256)
void gemm_bf16_kernel(const __bf16* __restrict__ A,   // [M][K] row-major
                      const __bf16* __restrict__ Bm,  // [batch][K][NTOK]
                      const float*  __restrict__ bias,// [M] or nullptr
                      float* __restrict__ Cm,         // [batch][M][NTOK]
                      int M, int K) {
  __shared__ __bf16 As[2 * 128 * 32]; // ping-pong [m][k]
  __shared__ __bf16 Bs[64 * 32];      // [n][k]  (transposed on store)
  int n0 = blockIdx.x * 64;
  int m0 = blockIdx.y * 128;
  const __bf16* Bg = Bm + (size_t)blockIdx.z * K * NTOK;
  float* Cg = Cm + (size_t)blockIdx.z * M * NTOK;

  int t = threadIdx.x, wave = t >> 5, lane = t & 31;
  int wm = wave & 3, wn = wave >> 2;          // wave tile coords
  int mrow = lane & 15, kh = (lane >= 16) ? 8 : 0;

  floatx8 acc[2][2] = {};
  int nIter = K >> 5;

#if HAVE_TDM
  if (wave == 0)
    tdm_load_tile_bf16((unsigned)(uintptr_t)&As[0], &A[(size_t)m0 * K],
                       32, 128, (unsigned)K);
#endif

  for (int it = 0; it < nIter; ++it) {
    int k0 = it << 5;
    __bf16* Ab = &As[(it & 1) * (128 * 32)];
#if HAVE_TDM
    if (wave == 0) {
      if (it + 1 < nIter) {
        tdm_load_tile_bf16((unsigned)(uintptr_t)&As[((it + 1) & 1) * (128 * 32)],
                           &A[(size_t)m0 * K + k0 + 32], 32, 128, (unsigned)K);
        __builtin_amdgcn_s_wait_tensorcnt(1);   // current buffer done, next in flight
      } else {
        __builtin_amdgcn_s_wait_tensorcnt(0);
      }
    }
#else
    {
      int row = t >> 1, ko = (t & 1) * 16;
      const __bf16* ga = &A[(size_t)(m0 + row) * K + k0 + ko];
      *(bf16x8*)&Ab[row * 32 + ko]     = *(const bf16x8*)(ga);
      *(bf16x8*)&Ab[row * 32 + ko + 8] = *(const bf16x8*)(ga + 8);
    }
#endif
    // stage B: one coalesced 16B global load per thread, transpose into LDS
    {
      int kk = t >> 3, nb = (t & 7) * 8;
      const __bf16* gb = &Bg[(size_t)(k0 + kk) * NTOK + n0 + nb];
      bf16x8 tmp = *(const bf16x8*)gb;
      if (k0 + 32 < K) __builtin_prefetch(gb + (size_t)32 * NTOK, 0, 0);
#pragma unroll
      for (int i = 0; i < 8; ++i) Bs[(nb + i) * 32 + kk] = tmp[i];
    }
    __syncthreads();

    bf16x16 af[2], bfr[2];
#pragma unroll
    for (int i = 0; i < 2; ++i) {
      af[i]  = load_frag(&Ab[(wm * 32 + i * 16 + mrow) * 32], kh);
      bfr[i] = load_frag(&Bs[(wn * 32 + i * 16 + mrow) * 32], kh);
    }
#pragma unroll
    for (int i = 0; i < 2; ++i)
#pragma unroll
      for (int j = 0; j < 2; ++j)
        acc[i][j] = __builtin_amdgcn_wmma_f32_16x16x32_bf16(
            false, af[i], false, bfr[j], (short)0, acc[i][j], false, false);
    __syncthreads();
  }

  // epilogue: C/D layout -> VGPR r holds M=r (lanes0-15) / r+8 (lanes16-31)
  int mofs = (lane >= 16) ? 8 : 0, ncol = lane & 15;
#pragma unroll
  for (int i = 0; i < 2; ++i)
#pragma unroll
    for (int j = 0; j < 2; ++j) {
      int gn = n0 + wn * 32 + j * 16 + ncol;
#pragma unroll
      for (int r = 0; r < 8; ++r) {
        int gm = m0 + wm * 32 + i * 16 + mofs + r;
        float v = acc[i][j][r];
        if (bias) v += bias[gm];
        Cg[(size_t)gm * NTOK + gn] = v;
      }
    }
}

// --- Kernel 4: per-(b,h,d) row max and 1/sum(exp) for k-softmax ----------
__global__ __launch_bounds__(256)
void kstats_kernel(const float* __restrict__ qkv,
                   float* __restrict__ kmax, float* __restrict__ kinv) {
  __shared__ float red[256];
  int row = blockIdx.x;              // bh*64 + d
  int bh = row >> 6;
  int b = bh >> 3, h = bh & 7, d = row & 63;
  const float* kr = qkv + ((size_t)b * O3 + HIDDEN + h * DHEAD + d) * NTOK;
  int t = threadIdx.x;
  float mx = -1e30f;
  for (int i = t; i < NTOK; i += 256) mx = fmaxf(mx, kr[i]);
  red[t] = mx; __syncthreads();
  for (int s = 128; s > 0; s >>= 1) { if (t < s) red[t] = fmaxf(red[t], red[t + s]); __syncthreads(); }
  mx = red[0]; __syncthreads();
  float sm = 0.f;
  for (int i = t; i < NTOK; i += 256) sm += __expf(kr[i] - mx);
  red[t] = sm; __syncthreads();
  for (int s = 128; s > 0; s >>= 1) { if (t < s) red[t] += red[t + s]; __syncthreads(); }
  if (t == 0) { kmax[row] = mx; kinv[row] = 1.f / red[0]; }
}

// --- Kernel 5: context[d,e] = sum_n softmax_k[d,n] * v[e,n] --------------
__global__ __launch_bounds__(256)
void ctx_kernel(const float* __restrict__ qkv, const float* __restrict__ kmax,
                const float* __restrict__ kinv, float* __restrict__ ctx) {
  __shared__ __bf16 Ks[64 * 32];   // [d][n-chunk]
  __shared__ __bf16 Vs[64 * 32];   // [e][n-chunk]
  int bh = blockIdx.x, b = bh >> 3, h = bh & 7;
  const float* kbase = qkv + ((size_t)b * O3 + HIDDEN + h * DHEAD) * NTOK;
  const float* vbase = qkv + ((size_t)b * O3 + 2 * HIDDEN + h * DHEAD) * NTOK;
  int t = threadIdx.x, wave = t >> 5, lane = t & 31;
  int dt = wave & 3, etb = (wave >> 2) * 2;
  int mrow = lane & 15, kh = (lane >= 16) ? 8 : 0;
  int sr = t >> 2, noff = (t & 3) * 8;     // staging: row 0..63, 8 n each
  float kmx = kmax[bh * 64 + sr], kin = kinv[bh * 64 + sr];

  floatx8 acc[2] = {};
  for (int n0 = 0; n0 < NTOK; n0 += 32) {
    bf16x8 ta, tv;
#pragma unroll
    for (int i = 0; i < 8; ++i) {
      float kv = kbase[(size_t)sr * NTOK + n0 + noff + i];
      ta[i] = (__bf16)(__expf(kv - kmx) * kin);
      tv[i] = (__bf16)vbase[(size_t)sr * NTOK + n0 + noff + i];
    }
    *(bf16x8*)&Ks[sr * 32 + noff] = ta;
    *(bf16x8*)&Vs[sr * 32 + noff] = tv;
    __syncthreads();
    bf16x16 af = load_frag(&Ks[(dt * 16 + mrow) * 32], kh);
#pragma unroll
    for (int j = 0; j < 2; ++j) {
      bf16x16 bfr = load_frag(&Vs[((etb + j) * 16 + mrow) * 32], kh);
      acc[j] = __builtin_amdgcn_wmma_f32_16x16x32_bf16(
          false, af, false, bfr, (short)0, acc[j], false, false);
    }
    __syncthreads();
  }
  int mofs = (lane >= 16) ? 8 : 0, ncol = lane & 15;
#pragma unroll
  for (int j = 0; j < 2; ++j)
#pragma unroll
    for (int r = 0; r < 8; ++r) {
      int dd = dt * 16 + mofs + r, ee = (etb + j) * 16 + ncol;
      ctx[(size_t)bh * 4096 + dd * 64 + ee] = acc[j][r];
    }
}

// --- Kernel 6: out[e,j] = sum_d ctx[d,e] * softmax_d(q)[d,j] * SCALE -----
__global__ __launch_bounds__(256)
void attn_kernel(const float* __restrict__ qkv, const float* __restrict__ ctx,
                 __bf16* __restrict__ attn) {
  __shared__ __bf16 Cts[64 * 64];   // [e][d]  (ctx transposed)
  __shared__ __bf16 Qs[128 * 64];   // [j][d]  (q softmax, transposed)
  int j0 = blockIdx.x * 128;
  int bh = blockIdx.y, b = bh >> 3, h = bh & 7;
  const float* qbase = qkv + ((size_t)b * O3 + h * DHEAD) * NTOK;
  const float* cbase = ctx + (size_t)bh * 4096;
  int t = threadIdx.x, wave = t >> 5, lane = t & 31;

  // stage ctx^T as bf16
#pragma unroll
  for (int i = 0; i < 16; ++i) {
    int idx = i * 256 + t;
    int dd = idx >> 6, ee = idx & 63;
    Cts[ee * 64 + dd] = (__bf16)cbase[idx];
  }
  // q softmax over d (64) per column; SCALE = 1/8
  if (t < 128) {
    int j = j0 + t;
    float mx = -1e30f;
    for (int dd = 0; dd < 64; ++dd) mx = fmaxf(mx, qbase[(size_t)dd * NTOK + j]);
    float sm = 0.f;
    for (int dd = 0; dd < 64; ++dd) sm += __expf(qbase[(size_t)dd * NTOK + j] - mx);
    float sc = 0.125f / sm;
    for (int dd = 0; dd < 64; ++dd)
      Qs[t * 64 + dd] = (__bf16)(__expf(qbase[(size_t)dd * NTOK + j] - mx) * sc);
  }
  __syncthreads();

  int et = wave & 3, jtb = (wave >> 2) * 4;
  int mrow = lane & 15, kh = (lane >= 16) ? 8 : 0;
  floatx8 acc[4] = {};
#pragma unroll
  for (int kc = 0; kc < 2; ++kc) {
    bf16x16 af = load_frag(&Cts[(et * 16 + mrow) * 64 + kc * 32], kh);
#pragma unroll
    for (int jj = 0; jj < 4; ++jj) {
      bf16x16 bfr = load_frag(&Qs[((jtb + jj) * 16 + mrow) * 64 + kc * 32], kh);
      acc[jj] = __builtin_amdgcn_wmma_f32_16x16x32_bf16(
          false, af, false, bfr, (short)0, acc[jj], false, false);
    }
  }
  int mofs = (lane >= 16) ? 8 : 0, ncol = lane & 15;
#pragma unroll
  for (int jj = 0; jj < 4; ++jj)
#pragma unroll
    for (int r = 0; r < 8; ++r) {
      int ee = et * 16 + mofs + r;
      int gj = j0 + (jtb + jj) * 16 + ncol;
      attn[((size_t)b * HIDDEN + h * DHEAD + ee) * NTOK + gj] = (__bf16)acc[jj][r];
    }
}

// --- Kernel 8: final RMS norm -------------------------------------------
__global__ __launch_bounds__(256)
void final_norm_kernel(const float* __restrict__ y, const float* __restrict__ g2,
                       float* __restrict__ out) {
  int gid = blockIdx.x * blockDim.x + threadIdx.x;
  int b = gid >> 12, j = gid & (NTOK - 1);
  const float* yp = y + (size_t)b * IN_CH * NTOK + j;
  float ss = 0.f;
  for (int c = 0; c < IN_CH; ++c) { float v = yp[(size_t)c * NTOK]; ss += v * v; }
  float s = 16.f / fmaxf(sqrtf(ss), 1e-12f);
  float* op = out + (size_t)b * IN_CH * NTOK + j;
  for (int c = 0; c < IN_CH; ++c) op[(size_t)c * NTOK] = yp[(size_t)c * NTOK] * s * g2[c];
}

// ---------------------------------------------------------------------------
extern "C" void kernel_launch(void* const* d_in, const int* in_sizes, int n_in,
                              void* d_out, int out_size, void* d_ws, size_t ws_size,
                              hipStream_t stream) {
  const float* x    = (const float*)d_in[0];
  const float* g1   = (const float*)d_in[1];
  const float* Wqkv = (const float*)d_in[2];
  const float* Wout = (const float*)d_in[3];
  const float* bout = (const float*)d_in[4];
  const float* g2   = (const float*)d_in[5];
  float* out = (float*)d_out;

  char* ws = (char*)d_ws;
  size_t off = 0;
  __bf16* xh    = (__bf16*)(ws + off); off += (size_t)NBATCH * IN_CH * NTOK * 2;   // 33.5 MB
  float*  qkv   = (float*) (ws + off); off += (size_t)NBATCH * O3 * NTOK * 4;      // 402 MB
  float*  kmaxb = (float*) (ws + off); off += (size_t)NBATCH * NHEADS * DHEAD * 4;
  float*  kinvb = (float*) (ws + off); off += (size_t)NBATCH * NHEADS * DHEAD * 4;
  float*  ctxb  = (float*) (ws + off); off += (size_t)NBATCH * NHEADS * DHEAD * DHEAD * 4;
  __bf16* attnb = (__bf16*)(ws + off); off += (size_t)NBATCH * HIDDEN * NTOK * 2;  // 67 MB
  float*  y     = (float*) (ws + off); off += (size_t)NBATCH * IN_CH * NTOK * 4;   // 67 MB
  __bf16* Wh    = (__bf16*)(ws + off); off += (size_t)O3 * IN_CH * 2;
  __bf16* Wouth = (__bf16*)(ws + off); off += (size_t)IN_CH * HIDDEN * 2;

  prep_x_kernel<<<NBATCH * NTOK / 256, 256, 0, stream>>>(x, xh);
  prep_w_kernel<<<(O3 * IN_CH + IN_CH * HIDDEN) / 256, 256, 0, stream>>>(Wqkv, g1, Wout, Wh, Wouth);
  gemm_bf16_kernel<<<dim3(NTOK / 64, O3 / 128, NBATCH), 256, 0, stream>>>(Wh, xh, nullptr, qkv, O3, IN_CH);
  kstats_kernel<<<NBATCH * NHEADS * DHEAD, 256, 0, stream>>>(qkv, kmaxb, kinvb);
  ctx_kernel<<<NBATCH * NHEADS, 256, 0, stream>>>(qkv, kmaxb, kinvb, ctxb);
  attn_kernel<<<dim3(NTOK / 128, NBATCH * NHEADS), 256, 0, stream>>>(qkv, ctxb, attnb);
  gemm_bf16_kernel<<<dim3(NTOK / 64, IN_CH / 128, NBATCH), 256, 0, stream>>>(Wouth, attnb, bout, y, IN_CH, HIDDEN);
  final_norm_kernel<<<NBATCH * NTOK / 256, 256, 0, stream>>>(y, g2, out);
}
